// SerializedPooling_86225763435012
// MI455X (gfx1250) — compile-verified, matrix operand order checked by reading, not run on hardware
//
#include <hip/hip_runtime.h>
#include <cstdint>
#include <cstddef>

// ============================================================================
// SerializedPooling for MI455X (gfx1250, wave32, WMMA).
//
// Heavy op: proj = feat @ W.T + b  (1M x 128 @ 128 x 256 = 68.7 GFLOP),
// fused with per-cluster segment_max.  Runs as split-bf16 (hi+lo) WMMA:
//   proj ~= Ahi*Bhi + Ahi*Blo + Alo*Bhi      (V_WMMA_F32_16X16X32_BF16)
// which keeps fp32-class accuracy at matrix-core rates; the stage is then
// HBM-bound (~512MB feat read @ 23.3 TB/s ~ 22us).
// ============================================================================

typedef __bf16 bf16;
typedef __attribute__((ext_vector_type(16))) __bf16 v16bf_t;
typedef __attribute__((ext_vector_type(8)))  float  v8f_t;

#define C_IN    128
#define C_OUT   256
#define POOL_D  1        // (STRIDE-1).bit_length() with STRIDE=2
#define LN_EPS  1e-5f
#define LDSK    136      // padded bf16 row stride in LDS (128 + 8)

// ---------------------------------------------------------------------------
// small device helpers
// ---------------------------------------------------------------------------
__device__ __forceinline__ unsigned mapOrderedF32(float f) {
  int i = __float_as_int(f);
  return (i >= 0) ? ((unsigned)i | 0x80000000u) : ~(unsigned)i;
}
__device__ __forceinline__ float unmapOrderedF32(unsigned u) {
  int i = (u & 0x80000000u) ? (int)(u ^ 0x80000000u) : ~(int)u;
  return __int_as_float(i);
}
__device__ __forceinline__ int pool_depth(const int* depthPtr) {
  return (POOL_D <= *depthPtr) ? POOL_D : 0;
}
__device__ __forceinline__ void bfsplit(float x, bf16& h, bf16& l) {
  h = (bf16)x;
  l = (bf16)(x - (float)h);
}

// ---------------------------------------------------------------------------
// radix sort (LSD, 8x8-bit passes, key=u64 sign-flipped, payload=int index)
// Unstable within a block -- safe here (see analysis notes).
// ---------------------------------------------------------------------------
#define RS_BLOCK 256
#define RS_ITEMS 4
#define RS_ELEMS (RS_BLOCK * RS_ITEMS)

__global__ void rs_hist(const unsigned long long* __restrict__ keys, int n, int shift,
                        unsigned* __restrict__ blockHist, int numBlocks) {
  __shared__ unsigned h[256];
  int t = threadIdx.x;
  h[t] = 0;
  __syncthreads();
  int base = blockIdx.x * RS_ELEMS;
  for (int it = 0; it < RS_ITEMS; ++it) {
    int i = base + it * RS_BLOCK + t;
    if (i < n) {
      unsigned d = (unsigned)((keys[i] >> shift) & 0xFF);
      atomicAdd(&h[d], 1u);
    }
  }
  __syncthreads();
  blockHist[(size_t)t * numBlocks + blockIdx.x] = h[t];
}

// one block per digit: exclusive scan of that digit's per-block counts
__global__ void rs_scan_blocks(unsigned* __restrict__ blockHist, int numBlocks,
                               unsigned* __restrict__ digitTotals) {
  __shared__ unsigned sh[RS_BLOCK];
  int d = blockIdx.x, t = threadIdx.x;
  unsigned* p = blockHist + (size_t)d * numBlocks;
  unsigned carry = 0;
  for (int base = 0; base < numBlocks; base += RS_BLOCK) {
    int i = base + t;
    unsigned v = (i < numBlocks) ? p[i] : 0u;
    sh[t] = v;
    __syncthreads();
    for (int off = 1; off < RS_BLOCK; off <<= 1) {
      unsigned a = (t >= off) ? sh[t - off] : 0u;
      __syncthreads();
      sh[t] += a;
      __syncthreads();
    }
    if (i < numBlocks) p[i] = sh[t] - v + carry;
    carry += sh[RS_BLOCK - 1];
    __syncthreads();
  }
  if (t == 0) digitTotals[d] = carry;
}

__global__ void rs_digit_base(const unsigned* __restrict__ totals,
                              unsigned* __restrict__ base) {
  __shared__ unsigned sh[256];
  int t = threadIdx.x;
  unsigned v = totals[t];
  sh[t] = v;
  __syncthreads();
  for (int off = 1; off < 256; off <<= 1) {
    unsigned a = (t >= off) ? sh[t - off] : 0u;
    __syncthreads();
    sh[t] += a;
    __syncthreads();
  }
  base[t] = sh[t] - v;
}

__global__ void rs_scatter(const unsigned long long* __restrict__ keysIn,
                           const int* __restrict__ idxIn, int n, int shift,
                           const unsigned* __restrict__ blockHist,
                           const unsigned* __restrict__ digitBase, int numBlocks,
                           unsigned long long* __restrict__ keysOut,
                           int* __restrict__ idxOut) {
  __shared__ unsigned cnt[256];
  int t = threadIdx.x;
  cnt[t] = 0;
  __syncthreads();
  int base = blockIdx.x * RS_ELEMS;
  for (int it = 0; it < RS_ITEMS; ++it) {
    int i = base + it * RS_BLOCK + t;
    if (i < n) {
      unsigned long long k = keysIn[i];
      unsigned d = (unsigned)((k >> shift) & 0xFF);
      unsigned r = atomicAdd(&cnt[d], 1u);
      unsigned pos = digitBase[d] + blockHist[(size_t)d * numBlocks + blockIdx.x] + r;
      keysOut[pos] = k;
      idxOut[pos]  = idxIn[i];
    }
  }
}

// ---------------------------------------------------------------------------
// key builders
// ---------------------------------------------------------------------------
__global__ void build_keys_c0(const long long* __restrict__ code0,
                              const int* __restrict__ depthPtr, int n,
                              unsigned long long* __restrict__ keys,
                              int* __restrict__ idx) {
  int i = blockIdx.x * blockDim.x + threadIdx.x;
  if (i < n) {
    int pd = pool_depth(depthPtr);
    long long c = code0[i] >> (3 * pd);
    keys[i] = ((unsigned long long)c) ^ 0x8000000000000000ull;  // signed order
    idx[i] = i;
  }
}

// composite (code << 20 | idx) reproduces stable argsort tie-breaking (M <= 2^20)
__global__ void build_keys_order(const long long* __restrict__ codeRow,
                                 const int* __restrict__ head,
                                 const int* __restrict__ depthPtr, int m,
                                 unsigned long long* __restrict__ keys,
                                 int* __restrict__ idx,
                                 float* __restrict__ outCodeRow) {
  int j = blockIdx.x * blockDim.x + threadIdx.x;
  if (j < m) {
    int pd = pool_depth(depthPtr);
    long long ch = codeRow[head[j]] >> (3 * pd);
    outCodeRow[j] = (float)ch;
    keys[j] = (((unsigned long long)ch) << 20) | (unsigned long long)(unsigned)j;
    idx[j] = j;
  }
}

// ---------------------------------------------------------------------------
// boundary flags + large prefix scan (for segment ids)
// ---------------------------------------------------------------------------
__global__ void seg_boundary(const unsigned long long* __restrict__ sk, int n,
                             int* __restrict__ flags) {
  int i = blockIdx.x * blockDim.x + threadIdx.x;
  if (i < n) flags[i] = (i == 0) || (sk[i] != sk[i - 1]);
}

#define SC_BLOCK 256
#define SC_ITEMS 8
#define SC_ELEMS (SC_BLOCK * SC_ITEMS)

__global__ void scan_block(const int* __restrict__ in, int n, int* __restrict__ out,
                           int* __restrict__ bsums) {
  __shared__ int sh[SC_BLOCK];
  int t = threadIdx.x;
  int base = blockIdx.x * SC_ELEMS + t * SC_ITEMS;
  int v[SC_ITEMS];
  int s = 0;
  for (int j = 0; j < SC_ITEMS; ++j) {
    int i = base + j;
    v[j] = (i < n) ? in[i] : 0;
    s += v[j];
  }
  sh[t] = s;
  __syncthreads();
  for (int off = 1; off < SC_BLOCK; off <<= 1) {
    int a = (t >= off) ? sh[t - off] : 0;
    __syncthreads();
    sh[t] += a;
    __syncthreads();
  }
  int run = sh[t] - s;
  if (t == SC_BLOCK - 1) bsums[blockIdx.x] = sh[SC_BLOCK - 1];
  for (int j = 0; j < SC_ITEMS; ++j) {
    run += v[j];
    int i = base + j;
    if (i < n) out[i] = run;
  }
}

__global__ void scan_sums(int* __restrict__ sums, int nb) {
  __shared__ int sh[SC_BLOCK];
  int t = threadIdx.x;
  int carry = 0;
  for (int base = 0; base < nb; base += SC_BLOCK) {
    int i = base + t;
    int v = (i < nb) ? sums[i] : 0;
    sh[t] = v;
    __syncthreads();
    for (int off = 1; off < SC_BLOCK; off <<= 1) {
      int a = (t >= off) ? sh[t - off] : 0;
      __syncthreads();
      sh[t] += a;
      __syncthreads();
    }
    if (i < nb) sums[i] = sh[t] - v + carry;
    carry += sh[SC_BLOCK - 1];
    __syncthreads();
  }
}

__global__ void scan_add(int* __restrict__ data, int n, const int* __restrict__ sums) {
  int i = blockIdx.x * blockDim.x + threadIdx.x;
  if (i < n) data[i] += sums[i / SC_ELEMS];
}

// cluster[orig] = seg; head[seg] = min(orig)  (== stable-argsort first hit)
__global__ void assign_cluster(const int* __restrict__ idxSorted,
                               const int* __restrict__ segIncl, int n,
                               int* __restrict__ cluster, int* __restrict__ head) {
  int i = blockIdx.x * blockDim.x + threadIdx.x;
  if (i < n) {
    int seg = segIncl[i] - 1;
    int idx = idxSorted[i];
    cluster[idx] = seg;
    atomicMin(&head[seg], idx);
  }
}

// ---------------------------------------------------------------------------
// W -> split bf16 (hi/lo)
// ---------------------------------------------------------------------------
__global__ void prep_w(const float* __restrict__ W, int total,
                       bf16* __restrict__ Whi, bf16* __restrict__ Wlo) {
  int i = blockIdx.x * blockDim.x + threadIdx.x;
  if (i < total) {
    bf16 h, l;
    bfsplit(W[i], h, l);
    Whi[i] = h;
    Wlo[i] = l;
  }
}

// ---------------------------------------------------------------------------
// fused WMMA GEMM (proj = feat @ W.T + b) + per-cluster segment_max epilogue
// workgroup tile: 128 points x 128 channels, 8 waves as 4(pts) x 2(ch)
// ---------------------------------------------------------------------------
__device__ __forceinline__ v16bf_t frag_a_lds(const bf16* arr, int m, int k0) {
  union { v16bf_t v; unsigned u[8]; } r;
  const unsigned* p = (const unsigned*)(arr + m * LDSK + k0);
  const unsigned* q = (const unsigned*)(arr + m * LDSK + k0 + 16);
#pragma unroll
  for (int j = 0; j < 4; ++j) { r.u[j] = p[j]; r.u[4 + j] = q[j]; }
  return r.v;
}
__device__ __forceinline__ v16bf_t frag_b_glb(const bf16* Wm, int o, int k0) {
  union { v16bf_t v; unsigned u[8]; } r;
  const unsigned* p = (const unsigned*)(Wm + (size_t)o * C_IN + k0);
  const unsigned* q = (const unsigned*)(Wm + (size_t)o * C_IN + k0 + 16);
#pragma unroll
  for (int j = 0; j < 4; ++j) { r.u[j] = p[j]; r.u[4 + j] = q[j]; }
  return r.v;
}

__global__ void __launch_bounds__(256)
gemm_segmax(const float* __restrict__ feat, const bf16* __restrict__ Whi,
            const bf16* __restrict__ Wlo, const float* __restrict__ bias,
            const int* __restrict__ cluster, int n,
            unsigned* __restrict__ featBits) {
  __shared__ bf16 lHi[128 * LDSK];
  __shared__ bf16 lLo[128 * LDSK];

  const int t = threadIdx.x;
  const int lane = t & 31;
  const int wave = t >> 5;
  const int ptBlock = blockIdx.x * 128;
  const int coBlock = blockIdx.y * 128;

  // prefetch next point tile (speculative; dropped if OOB)
  __builtin_prefetch((const void*)(feat + (size_t)(ptBlock + 128) * C_IN), 0, 1);

  // ---- stage 128x128 f32 feat tile into LDS as split bf16 (hi/lo) ----
  {
    const float4* f4 = (const float4*)feat;
    for (int it = 0; it < 16; ++it) {
      int idx = it * 256 + t;        // 0..4095 float4 slots
      int row = idx >> 5;            // 32 float4 per 128-float row
      int c4  = idx & 31;
      int gp  = ptBlock + row;
      int src = (gp < n) ? gp : (n - 1);
      float4 v = f4[(size_t)src * (C_IN / 4) + c4];
      int lb = row * LDSK + c4 * 4;
      bf16 h, l;
      bfsplit(v.x, h, l); lHi[lb + 0] = h; lLo[lb + 0] = l;
      bfsplit(v.y, h, l); lHi[lb + 1] = h; lLo[lb + 1] = l;
      bfsplit(v.z, h, l); lHi[lb + 2] = h; lLo[lb + 2] = l;
      bfsplit(v.w, h, l); lHi[lb + 3] = h; lLo[lb + 3] = l;
    }
  }
  __syncthreads();

  const int mcol   = lane & 15;
  const int khalf  = (lane & 16) ? 8 : 0;   // A/B k-half per 16-bit WMMA layout
  const int ptWave = (wave & 3) * 32;       // 32 points per wave
  const int coWave = (wave >> 2) * 64;      // 64 channels per wave (local)

  v8f_t acc[2][4] = {};

  // 3 precision passes: Ahi*Bhi, Ahi*Blo, Alo*Bhi
  for (int pass = 0; pass < 3; ++pass) {
    const bf16* Aarr = (pass == 2) ? lLo : lHi;
    const bf16* Barr = (pass == 1) ? Wlo : Whi;
#pragma unroll
    for (int ks = 0; ks < 4; ++ks) {
      const int kb = ks * 32 + khalf;
      v16bf_t a0 = frag_a_lds(Aarr, ptWave + mcol, kb);
      v16bf_t a1 = frag_a_lds(Aarr, ptWave + 16 + mcol, kb);
#pragma unroll
      for (int tj = 0; tj < 4; ++tj) {
        const int o = coBlock + coWave + tj * 16 + mcol;
        v16bf_t bq = frag_b_glb(Barr, o, kb);
        acc[0][tj] = __builtin_amdgcn_wmma_f32_16x16x32_bf16(
            false, a0, false, bq, (short)0, acc[0][tj], false, false);
        acc[1][tj] = __builtin_amdgcn_wmma_f32_16x16x32_bf16(
            false, a1, false, bq, (short)0, acc[1][tj], false, false);
      }
    }
  }

  // ---- epilogue: +bias, order-preserving-int atomicMax into pooled rows ----
  const int rowAdd = (lane & 16) ? 8 : 0;   // C/D layout: lanes16-31 hold M=r+8
#pragma unroll
  for (int ti = 0; ti < 2; ++ti) {
#pragma unroll
    for (int tj = 0; tj < 4; ++tj) {
      const int o = coBlock + coWave + tj * 16 + mcol;
      const float bv = bias[o];
#pragma unroll
      for (int r = 0; r < 8; ++r) {
        const int m = ptBlock + ptWave + ti * 16 + r + rowAdd;
        if (m < n) {
          float v = acc[ti][tj][r] + bv;
          atomicMax(&featBits[(size_t)cluster[m] * C_OUT + o], mapOrderedF32(v));
        }
      }
    }
  }
}

// ---------------------------------------------------------------------------
// segment coord mean, head-side gathers, order writes, cluster dump
// ---------------------------------------------------------------------------
__global__ void coord_acc(const float* __restrict__ coord, const int* __restrict__ cluster,
                          int n, float* __restrict__ coordOut, float* __restrict__ counts) {
  int i = blockIdx.x * blockDim.x + threadIdx.x;
  if (i < n) {
    int c = cluster[i];
    atomicAdd(&coordOut[(size_t)c * 3 + 0], coord[(size_t)i * 3 + 0]);
    atomicAdd(&coordOut[(size_t)c * 3 + 1], coord[(size_t)i * 3 + 1]);
    atomicAdd(&coordOut[(size_t)c * 3 + 2], coord[(size_t)i * 3 + 2]);
    atomicAdd(&counts[c], 1.0f);
  }
}

__global__ void coord_div(float* __restrict__ coordOut, const float* __restrict__ counts,
                          int m3) {
  int i = blockIdx.x * blockDim.x + threadIdx.x;
  if (i < m3) coordOut[i] /= counts[i / 3];
}

__global__ void head_side(const int* __restrict__ head, const int* __restrict__ gridc,
                          const long long* __restrict__ batch,
                          const int* __restrict__ depthPtr, int m,
                          float* __restrict__ outGrid, float* __restrict__ outBatch) {
  int j = blockIdx.x * blockDim.x + threadIdx.x;
  if (j < m) {
    int pd = pool_depth(depthPtr);
    int h = head[j];
    outGrid[(size_t)j * 3 + 0] = (float)(gridc[(size_t)h * 3 + 0] >> pd);
    outGrid[(size_t)j * 3 + 1] = (float)(gridc[(size_t)h * 3 + 1] >> pd);
    outGrid[(size_t)j * 3 + 2] = (float)(gridc[(size_t)h * 3 + 2] >> pd);
    outBatch[j] = (float)batch[h];
  }
}

__global__ void write_order(const int* __restrict__ idxSorted, int m,
                            float* __restrict__ outOrder, float* __restrict__ outInv) {
  int i = blockIdx.x * blockDim.x + threadIdx.x;
  if (i < m) {
    int p = idxSorted[i];
    outOrder[i] = (float)p;
    outInv[p] = (float)i;
  }
}

__global__ void cluster_to_float(const int* __restrict__ cluster, int n,
                                 float* __restrict__ out) {
  int i = blockIdx.x * blockDim.x + threadIdx.x;
  if (i < n) out[i] = (float)cluster[i];
}

// ---------------------------------------------------------------------------
// LayerNorm (population var) + exact GELU, in place over pooled rows
// ---------------------------------------------------------------------------
__global__ void ln_gelu(unsigned* __restrict__ xbits, float* __restrict__ xout,
                        const float* __restrict__ gamma, const float* __restrict__ beta,
                        int m) {
  __shared__ float red[C_OUT];
  int row = blockIdx.x;
  int t = threadIdx.x;
  float v = unmapOrderedF32(xbits[(size_t)row * C_OUT + t]);
  red[t] = v;
  __syncthreads();
  for (int s = C_OUT / 2; s > 0; s >>= 1) {
    if (t < s) red[t] += red[t + s];
    __syncthreads();
  }
  float mu = red[0] * (1.0f / C_OUT);
  __syncthreads();
  float d = v - mu;
  red[t] = d * d;
  __syncthreads();
  for (int s = C_OUT / 2; s > 0; s >>= 1) {
    if (t < s) red[t] += red[t + s];
    __syncthreads();
  }
  float var = red[0] * (1.0f / C_OUT);
  float y = d * rsqrtf(var + LN_EPS) * gamma[t] + beta[t];
  float g = 0.5f * y * (1.0f + erff(y * 0.70710678118654752f));
  xout[(size_t)row * C_OUT + t] = g;
}

// ---------------------------------------------------------------------------
// host-side drivers
// ---------------------------------------------------------------------------
static void run_radix_sort(hipStream_t s, unsigned long long* kA, int* iA,
                           unsigned long long* kB, int* iB, unsigned* blockHist,
                           unsigned* digitTotals, unsigned* digitBase, int n) {
  int nb = (n + RS_ELEMS - 1) / RS_ELEMS;
  unsigned long long *ki = kA, *ko = kB;
  int *ii = iA, *io = iB;
  for (int p = 0; p < 8; ++p) {
    int shift = p * 8;
    rs_hist<<<nb, RS_BLOCK, 0, s>>>(ki, n, shift, blockHist, nb);
    rs_scan_blocks<<<256, RS_BLOCK, 0, s>>>(blockHist, nb, digitTotals);
    rs_digit_base<<<1, 256, 0, s>>>(digitTotals, digitBase);
    rs_scatter<<<nb, RS_BLOCK, 0, s>>>(ki, ii, n, shift, blockHist, digitBase, nb, ko, io);
    unsigned long long* tk = ki; ki = ko; ko = tk;
    int* ti = ii; ii = io; io = ti;
  }
  // 8 passes (even) -> results back in kA / iA
}

static void run_scan(hipStream_t s, const int* in, int* out, int* bsums, int n) {
  int nb = (n + SC_ELEMS - 1) / SC_ELEMS;
  scan_block<<<nb, SC_BLOCK, 0, s>>>(in, n, out, bsums);
  scan_sums<<<1, SC_BLOCK, 0, s>>>(bsums, nb);
  scan_add<<<(n + 255) / 256, 256, 0, s>>>(out, n, bsums);
}

extern "C" void kernel_launch(void* const* d_in, const int* in_sizes, int n_in,
                              void* d_out, int out_size, void* d_ws, size_t ws_size,
                              hipStream_t stream) {
  const float*     feat     = (const float*)d_in[0];
  const float*     coord    = (const float*)d_in[1];
  const int*       gridc    = (const int*)d_in[2];
  const long long* code     = (const long long*)d_in[3];   // [2][N] int64
  const long long* batch    = (const long long*)d_in[4];   // [N] int64
  const int*       depthPtr = (const int*)d_in[5];         // scalar (low word)
  const float*     W        = (const float*)d_in[6];
  const float*     bias     = (const float*)d_in[7];
  const float*     gamma    = (const float*)d_in[8];
  const float*     beta     = (const float*)d_in[9];
  (void)n_in; (void)ws_size;

  const int N = in_sizes[4];
  // out_size = M*(256+3+3+2+2+2+1) + N = 269*M + N   -> recover M on host
  const int M = (int)(((long long)out_size - (long long)N) / 269);

  float* out = (float*)d_out;
  float* out_x       = out;
  float* out_coord   = out_x     + (size_t)M * C_OUT;
  float* out_grid    = out_coord + (size_t)M * 3;
  float* out_code    = out_grid  + (size_t)M * 3;   // [2][M]
  float* out_order   = out_code  + (size_t)2 * M;   // [2][M]
  float* out_inv     = out_order + (size_t)2 * M;   // [2][M]
  float* out_cluster = out_inv   + (size_t)2 * M;   // [N]
  float* out_batchp  = out_cluster + (size_t)N;     // [M]
  unsigned* featBits = (unsigned*)out_x;            // pooled max lives in-place

  // ---- workspace carve ----
  char* p = (char*)d_ws;
  auto take = [&p](size_t bytes) -> char* {
    char* q = (char*)(((uintptr_t)p + 255) & ~(uintptr_t)255);
    p = q + bytes;
    return q;
  };
  int nbHist = (N + RS_ELEMS - 1) / RS_ELEMS;
  unsigned long long* keysA = (unsigned long long*)take((size_t)N * 8);
  unsigned long long* keysB = (unsigned long long*)take((size_t)N * 8);
  int*      idxA       = (int*)take((size_t)N * 4);
  int*      idxB       = (int*)take((size_t)N * 4);
  unsigned* blockHist  = (unsigned*)take((size_t)256 * nbHist * 4);
  unsigned* digitTot   = (unsigned*)take(256 * 4);
  unsigned* digitBase  = (unsigned*)take(256 * 4);
  int*      flags      = (int*)take((size_t)N * 4);
  int*      segIncl    = (int*)take((size_t)N * 4);
  int*      bsums      = (int*)take((size_t)((N + SC_ELEMS - 1) / SC_ELEMS) * 4 + 256);
  int*      cluster    = (int*)take((size_t)N * 4);
  int*      head       = (int*)take((size_t)N * 4);
  float*    counts     = (float*)take((size_t)N * 4);
  bf16*     Whi        = (bf16*)take((size_t)C_OUT * C_IN * 2);
  bf16*     Wlo        = (bf16*)take((size_t)C_OUT * C_IN * 2);

  // ---- init (graph-capture-safe memsets) ----
  hipMemsetAsync(featBits, 0, (size_t)M * C_OUT * 4, stream);       // mapped -inf
  hipMemsetAsync(out_coord, 0, (size_t)M * 3 * 4, stream);
  hipMemsetAsync(counts, 0, (size_t)M * 4, stream);
  hipMemsetAsync(head, 0x7F, (size_t)N * 4, stream);                // ~INT_MAX

  // ---- W -> split bf16 ----
  prep_w<<<(C_OUT * C_IN + 255) / 256, 256, 0, stream>>>(W, C_OUT * C_IN, Whi, Wlo);

  // ---- sort c0, build clusters + heads ----
  build_keys_c0<<<(N + 255) / 256, 256, 0, stream>>>(code, depthPtr, N, keysA, idxA);
  run_radix_sort(stream, keysA, idxA, keysB, idxB, blockHist, digitTot, digitBase, N);
  seg_boundary<<<(N + 255) / 256, 256, 0, stream>>>(keysA, N, flags);
  run_scan(stream, flags, segIncl, bsums, N);
  assign_cluster<<<(N + 255) / 256, 256, 0, stream>>>(idxA, segIncl, N, cluster, head);

  // ---- fused WMMA GEMM + segment_max ----
  dim3 gg((N + 127) / 128, C_OUT / 128);
  gemm_segmax<<<gg, 256, 0, stream>>>(feat, Whi, Wlo, bias, cluster, N, featBits);

  // ---- segment coord mean ----
  coord_acc<<<(N + 255) / 256, 256, 0, stream>>>(coord, cluster, N, out_coord, counts);
  coord_div<<<(M * 3 + 255) / 256, 256, 0, stream>>>(out_coord, counts, M * 3);

  // ---- head-side gathers ----
  head_side<<<(M + 255) / 256, 256, 0, stream>>>(head, gridc, batch, depthPtr, M,
                                                 out_grid, out_batchp);

  // ---- per-row argsort of code_h (stable via composite keys) ----
  for (int r = 0; r < 2; ++r) {
    build_keys_order<<<(M + 255) / 256, 256, 0, stream>>>(
        code + (size_t)r * N, head, depthPtr, M, keysA, idxA, out_code + (size_t)r * M);
    run_radix_sort(stream, keysA, idxA, keysB, idxB, blockHist, digitTot, digitBase, M);
    write_order<<<(M + 255) / 256, 256, 0, stream>>>(
        idxA, M, out_order + (size_t)r * M, out_inv + (size_t)r * M);
  }

  // ---- cluster dump ----
  cluster_to_float<<<(N + 255) / 256, 256, 0, stream>>>(cluster, N, out_cluster);

  // ---- LayerNorm + exact GELU (in place over pooled rows) ----
  ln_gelu<<<M, C_OUT, 0, stream>>>(featBits, out_x, gamma, beta, M);
}
